// DNN_34497177321482
// MI455X (gfx1250) — matrix operand order repaired and checked by reading, not compiled, not run on hardware
//
#include <hip/hip_runtime.h>

// Problem constants (match reference)
#define B_    512
#define L_    200
#define D_    256
#define K_    2048
#define NROWS (B_ * L_)       // 102400
#define DP2   (D_ + 8)        // LDS row stride in bf16 elements (264: pad, keeps 16B align)
#define MROWS 32              // rows per block (two 16-row WMMA tiles)

typedef float  v8f   __attribute__((ext_vector_type(8)));
typedef __bf16 v8bf  __attribute__((ext_vector_type(8)));
typedef __bf16 v16bf __attribute__((ext_vector_type(16)));

// ---- optional CDNA5 async global->LDS path (ASYNCcnt), guarded ----
#if defined(__has_builtin)
#if __has_builtin(__builtin_amdgcn_global_load_async_to_lds_b128) && \
    __has_builtin(__builtin_amdgcn_s_wait_asynccnt)
#define HAVE_ASYNC_LDS 1
#endif
#endif

// Pointer types matching the builtin's signature exactly as diagnosed:
//   param0: 'int __vector_size__(16) __device__ *'  (global, AS1)
//   param1: LDS destination (AS3, 32-bit pointer)
typedef int v4i_t __attribute__((vector_size(16)));
typedef __attribute__((address_space(1))) v4i_t* glb_b128_p;
typedef __attribute__((address_space(3))) v4i_t* lds_b128_p;

__device__ __forceinline__ glb_b128_p to_glb128(const void* p) {
    return (glb_b128_p)(unsigned long long)p;
}
__device__ __forceinline__ lds_b128_p to_lds128(void* p) {
    // generic LDS addr: [31:0] is the raw LDS byte offset (aperture mapping)
    return (lds_b128_p)(unsigned int)(unsigned long long)p;
}

// ---------------------------------------------------------------------------
// Kernel 0: cnorm[k] = ||code_book[k]||^2   (exact fp32)
// ---------------------------------------------------------------------------
__global__ void cnorm_kernel(const float* __restrict__ cb, float* __restrict__ cnorm) {
    int k = blockIdx.x * blockDim.x + threadIdx.x;
    if (k < K_) {
        float s = 0.0f;
        const float* row = cb + (size_t)k * D_;
        #pragma unroll 8
        for (int d = 0; d < D_; ++d) { float v = row[d]; s += v * v; }
        cnorm[k] = s;
    }
}

// ---------------------------------------------------------------------------
// Kernel 0b: split codebook into bf16 hi/lo once (error-compensated GEMM input)
// ---------------------------------------------------------------------------
__global__ __launch_bounds__(256)
void cb_split_kernel(const float* __restrict__ cb,
                     __bf16* __restrict__ cbhi, __bf16* __restrict__ cblo) {
    size_t i = (size_t)blockIdx.x * blockDim.x + threadIdx.x;   // K_*D_ threads
    float x = cb[i];
    __bf16 h = (__bf16)x;
    cbhi[i] = h;
    cblo[i] = (__bf16)(x - (float)h);
}

// ---------------------------------------------------------------------------
// Kernel 1 (main): split-bf16 WMMA distance GEMM + running argmin.
// Block = 32 rows (two 16-row tiles per wave, shared B fragment), 4 waves
// each own 16 columns of a 64-col LDS codebook tile.
// dot = AhiBhi + AhiBlo + AloBhi per 32-K chunk (2 independent acc chains).
// ---------------------------------------------------------------------------
__global__ __launch_bounds__(128)
void vq_argmin_kernel(const int*    __restrict__ ids,
                      const int*    __restrict__ masks,
                      const float*  __restrict__ emb,
                      const __bf16* __restrict__ cbhi,
                      const __bf16* __restrict__ cblo,
                      const float*  __restrict__ cnorm,
                      int*          __restrict__ argidx)
{
    __shared__ __bf16 sAhi[MROWS * DP2];
    __shared__ __bf16 sAlo[MROWS * DP2];
    __shared__ __bf16 sBhi[64 * DP2];
    __shared__ __bf16 sBlo[64 * DP2];
    __shared__ float  redV[4 * MROWS];
    __shared__ int    redI[4 * MROWS];

    const int tid   = threadIdx.x;     // 0..127
    const int wave  = tid >> 5;        // 0..3
    const int lane  = tid & 31;        // wave32
    const int half  = lane >> 4;
    const int l16   = lane & 15;
    const int rowBase = blockIdx.x * MROWS;

    // ---- Stage A tile: 32 rows x 256, scaled by -2*mask, split hi/lo ----
    for (int v = tid; v < MROWS * (D_ / 8); v += 128) {   // 1024 8-elt chunks
        int r  = v >> 5;               // 32 chunks per row
        int c8 = v & 31;
        int flat = rowBase + r;
        int id   = ids[flat];
        float m  = (masks[flat] >= 1) ? -2.0f : 0.0f;
        const float4* src = (const float4*)(emb + (size_t)id * D_) + c8 * 2;
        float4 a0 = src[0], a1 = src[1];
        float xs[8] = {a0.x, a0.y, a0.z, a0.w, a1.x, a1.y, a1.z, a1.w};
        v8bf hi, lo;
        #pragma unroll
        for (int i = 0; i < 8; ++i) {
            float x = xs[i] * m;
            __bf16 h = (__bf16)x;
            hi[i] = h;
            lo[i] = (__bf16)(x - (float)h);
        }
        *(v8bf*)(&sAhi[r * DP2 + c8 * 8]) = hi;
        *(v8bf*)(&sAlo[r * DP2 + c8 * 8]) = lo;
    }

    float bestV[16];
    int   bestI[16];
    #pragma unroll
    for (int j = 0; j < 16; ++j) { bestV[j] = 3.4e38f; bestI[j] = 0; }

    for (int kc = 0; kc < K_; kc += 64) {
        __syncthreads();
        // ---- Stage B tile (64 codes), hi/lo bf16 ----
#ifdef HAVE_ASYNC_LDS
        for (int v = tid; v < 64 * (D_ / 8); v += 128) {   // 2048 16B chunks
            int n  = v >> 5;
            int c8 = v & 31;
            size_t g = (size_t)(kc + n) * D_ + c8 * 8;
            int    l = n * DP2 + c8 * 8;
            __builtin_amdgcn_global_load_async_to_lds_b128(to_glb128(cbhi + g),
                                                           to_lds128(&sBhi[l]), 0, 0);
            __builtin_amdgcn_global_load_async_to_lds_b128(to_glb128(cblo + g),
                                                           to_lds128(&sBlo[l]), 0, 0);
        }
        __builtin_amdgcn_s_wait_asynccnt(0);
#else
        for (int v = tid; v < 64 * (D_ / 8); v += 128) {
            int n  = v >> 5;
            int c8 = v & 31;
            size_t g = (size_t)(kc + n) * D_ + c8 * 8;
            *(v8bf*)(&sBhi[n * DP2 + c8 * 8]) = *(const v8bf*)(cbhi + g);
            *(v8bf*)(&sBlo[n * DP2 + c8 * 8]) = *(const v8bf*)(cblo + g);
        }
#endif
        if (kc + 64 < K_) {  // hint next tile toward cache (global_prefetch_b8)
            __builtin_prefetch(cbhi + (size_t)(kc + 64) * D_ + tid * 8, 0, 1);
            __builtin_prefetch(cblo + (size_t)(kc + 64) * D_ + tid * 8, 0, 1);
        }
        __syncthreads();

        const int colBase = kc + wave * 16;
        const __bf16* aHi0 = &sAhi[l16 * DP2];          // tile0: rows 0..15
        const __bf16* aLo0 = &sAlo[l16 * DP2];
        const __bf16* aHi1 = &sAhi[(16 + l16) * DP2];   // tile1: rows 16..31
        const __bf16* aLo1 = &sAlo[(16 + l16) * DP2];
        const __bf16* bHiR = &sBhi[(wave * 16 + l16) * DP2];
        const __bf16* bLoR = &sBlo[(wave * 16 + l16) * DP2];

        v8f acc0 = {}, acc1 = {};
        #pragma unroll
        for (int k0 = 0; k0 < D_; k0 += 32) {
            // A frag (16-bit 16x32): half 0 -> K {0..7,16..23}, half 1 -> {8..15,24..31}
            v16bf ahi0 = __builtin_shufflevector(
                *(const v8bf*)(aHi0 + k0 + half * 8),
                *(const v8bf*)(aHi0 + k0 + 16 + half * 8),
                0,1,2,3,4,5,6,7,8,9,10,11,12,13,14,15);
            v16bf alo0 = __builtin_shufflevector(
                *(const v8bf*)(aLo0 + k0 + half * 8),
                *(const v8bf*)(aLo0 + k0 + 16 + half * 8),
                0,1,2,3,4,5,6,7,8,9,10,11,12,13,14,15);
            v16bf ahi1 = __builtin_shufflevector(
                *(const v8bf*)(aHi1 + k0 + half * 8),
                *(const v8bf*)(aHi1 + k0 + 16 + half * 8),
                0,1,2,3,4,5,6,7,8,9,10,11,12,13,14,15);
            v16bf alo1 = __builtin_shufflevector(
                *(const v8bf*)(aLo1 + k0 + half * 8),
                *(const v8bf*)(aLo1 + k0 + 16 + half * 8),
                0,1,2,3,4,5,6,7,8,9,10,11,12,13,14,15);
            // B frag (32x16): half 0 -> K {0..7,16..23}, half 1 -> {8..15,24..31}
            v16bf bhi = __builtin_shufflevector(
                *(const v8bf*)(bHiR + k0 + half * 8),
                *(const v8bf*)(bHiR + k0 + 16 + half * 8),
                0,1,2,3,4,5,6,7,8,9,10,11,12,13,14,15);
            v16bf blo = __builtin_shufflevector(
                *(const v8bf*)(bLoR + k0 + half * 8),
                *(const v8bf*)(bLoR + k0 + 16 + half * 8),
                0,1,2,3,4,5,6,7,8,9,10,11,12,13,14,15);

            acc0 = __builtin_amdgcn_wmma_f32_16x16x32_bf16(false, ahi0, false, bhi,
                                                           (short)0, acc0, false, false);
            acc1 = __builtin_amdgcn_wmma_f32_16x16x32_bf16(false, ahi1, false, bhi,
                                                           (short)0, acc1, false, false);
            acc0 = __builtin_amdgcn_wmma_f32_16x16x32_bf16(false, ahi0, false, blo,
                                                           (short)0, acc0, false, false);
            acc1 = __builtin_amdgcn_wmma_f32_16x16x32_bf16(false, ahi1, false, blo,
                                                           (short)0, acc1, false, false);
            acc0 = __builtin_amdgcn_wmma_f32_16x16x32_bf16(false, alo0, false, bhi,
                                                           (short)0, acc0, false, false);
            acc1 = __builtin_amdgcn_wmma_f32_16x16x32_bf16(false, alo1, false, bhi,
                                                           (short)0, acc1, false, false);
        }

        const int col = colBase + l16;
        const float cn = cnorm[col];
        #pragma unroll
        for (int j = 0; j < 8; ++j) {     // C layout: VGPR j -> M = j + half*8
            float s0 = acc0[j] + cn;
            if (s0 < bestV[j])     { bestV[j] = s0;     bestI[j] = col; }
            float s1 = acc1[j] + cn;
            if (s1 < bestV[j + 8]) { bestV[j + 8] = s1; bestI[j + 8] = col; }
        }
    }

    // ---- argmin across the 16 lanes of each half (first-index tie-break) ----
    #pragma unroll
    for (int j = 0; j < 16; ++j) {
        float v = bestV[j]; int i = bestI[j];
        #pragma unroll
        for (int off = 1; off < 16; off <<= 1) {
            float ov = __shfl_xor(v, off, 32);
            int   oi = __shfl_xor(i, off, 32);
            if (ov < v || (ov == v && oi < i)) { v = ov; i = oi; }
        }
        bestV[j] = v; bestI[j] = i;
    }
    if (l16 == 0) {
        #pragma unroll
        for (int j = 0; j < 8; ++j) {
            int m0 = j + half * 8;           // tile0 row
            int m1 = 16 + j + half * 8;      // tile1 row
            redV[wave * MROWS + m0] = bestV[j];
            redI[wave * MROWS + m0] = bestI[j];
            redV[wave * MROWS + m1] = bestV[j + 8];
            redI[wave * MROWS + m1] = bestI[j + 8];
        }
    }
    __syncthreads();
    if (tid < MROWS) {
        float v = redV[tid]; int i = redI[tid];
        #pragma unroll
        for (int w = 1; w < 4; ++w) {
            float ov = redV[w * MROWS + tid]; int oi = redI[w * MROWS + tid];
            if (ov < v || (ov == v && oi < i)) { v = ov; i = oi; }
        }
        argidx[rowBase + tid] = i;
    }
}

// ---------------------------------------------------------------------------
// Kernel 2: masked history mean -> xcat[b][256+d]   (exact fp32)
// ---------------------------------------------------------------------------
__global__ __launch_bounds__(256)
void hist_mean_kernel(const int* __restrict__ ids, const int* __restrict__ masks,
                      const float* __restrict__ emb, float* __restrict__ xcat)
{
    int b = blockIdx.x, d = threadIdx.x;
    float acc = 0.0f, cnt = 0.0f;
    for (int l = 0; l < L_; ++l) {
        int flat = b * L_ + l;
        if (masks[flat] >= 1) {
            cnt += 1.0f;
            acc += emb[(size_t)ids[flat] * D_ + d];
        }
    }
    xcat[b * (2 * D_) + D_ + d] = acc / (cnt + 1e-9f);
}

// ---------------------------------------------------------------------------
// Kernel 3: vq mean (gather chosen codes, sum ALL L rows) -> xcat[b][d]
// ---------------------------------------------------------------------------
__global__ __launch_bounds__(256)
void vq_mean_kernel(const int* __restrict__ masks, const int* __restrict__ argidx,
                    const float* __restrict__ cb, float* __restrict__ xcat)
{
    int b = blockIdx.x, d = threadIdx.x;
    float acc = 0.0f, cnt = 0.0f;
    for (int l = 0; l < L_; ++l) {
        int flat = b * L_ + l;
        cnt += (masks[flat] >= 1) ? 1.0f : 0.0f;
        acc += cb[(size_t)argidx[flat] * D_ + d];
    }
    xcat[b * (2 * D_) + d] = acc / cnt;   // reference: no eps here
}

// ---------------------------------------------------------------------------
// Kernel 4: out = xcat @ W_enc + b_enc   ([512,512] x [512,256])
// ---------------------------------------------------------------------------
__global__ __launch_bounds__(256)
void dense_kernel(const float* __restrict__ xcat, const float* __restrict__ W,
                  const float* __restrict__ bias, float* __restrict__ out)
{
    __shared__ float sx[2 * D_];
    int b = blockIdx.x, j = threadIdx.x;
    sx[j]      = xcat[b * (2 * D_) + j];
    sx[j + D_] = xcat[b * (2 * D_) + D_ + j];
    __syncthreads();
    float acc = bias[j];
    #pragma unroll 8
    for (int i = 0; i < 2 * D_; ++i)
        acc += sx[i] * W[i * D_ + j];
    out[b * D_ + j] = acc;
}

// ---------------------------------------------------------------------------
extern "C" void kernel_launch(void* const* d_in, const int* in_sizes, int n_in,
                              void* d_out, int out_size, void* d_ws, size_t ws_size,
                              hipStream_t stream) {
    const int*   ids   = (const int*)  d_in[0];   // [B,L]
    const int*   masks = (const int*)  d_in[1];   // [B,L]
    const float* emb   = (const float*)d_in[2];   // [V,D]
    const float* cb    = (const float*)d_in[3];   // [K,D]
    const float* W     = (const float*)d_in[4];   // [2D,D]
    const float* bias  = (const float*)d_in[5];   // [D]
    float* out = (float*)d_out;                   // [B,D]

    // Workspace layout: cnorm | argidx | xcat | cbhi | cblo
    float*  ws_cnorm = (float*)d_ws;                         // K_ floats
    int*    ws_idx   = (int*)(ws_cnorm + K_);                // NROWS ints
    float*  ws_xcat  = (float*)(ws_idx + NROWS);             // B_*2D floats
    __bf16* ws_cbhi  = (__bf16*)(ws_xcat + (size_t)B_ * 2 * D_);   // K_*D_ bf16
    __bf16* ws_cblo  = ws_cbhi + (size_t)K_ * D_;                  // K_*D_ bf16

    cnorm_kernel    <<<K_ / 256,          256, 0, stream>>>(cb, ws_cnorm);
    cb_split_kernel <<<(K_ * D_) / 256,   256, 0, stream>>>(cb, ws_cbhi, ws_cblo);
    vq_argmin_kernel<<<NROWS / MROWS,     128, 0, stream>>>(ids, masks, emb,
                                                            ws_cbhi, ws_cblo,
                                                            ws_cnorm, ws_idx);
    hist_mean_kernel<<<B_,                256, 0, stream>>>(ids, masks, emb, ws_xcat);
    vq_mean_kernel  <<<B_,                256, 0, stream>>>(masks, ws_idx, cb, ws_xcat);
    dense_kernel    <<<B_,                256, 0, stream>>>(ws_xcat, W, bias, out);
}
